// Block_54185307407114
// MI455X (gfx1250) — matrix operand order repaired and checked by reading, not compiled, not run on hardware
//
#include <hip/hip_runtime.h>

// ---------------------------------------------------------------- constants
constexpr int Bb_ = 8, T_ = 1024, C_ = 1024, H_ = 16, HS_ = 64, TT_ = 1024;
constexpr int BT_ = Bb_ * T_;            // 8192 rows
constexpr int C3_ = 3 * C_;              // 3072

typedef _Float16 half_t;
typedef __attribute__((ext_vector_type(16))) _Float16 v16h;
typedef __attribute__((ext_vector_type(8)))  _Float16 v8h;
typedef __attribute__((ext_vector_type(8)))  float    v8f;

// ---------------------------------------------------------------- workspace map (bytes)
constexpr size_t MB = 1ull << 20;
constexpr size_t OFF_WK    = 0;          // [C,C]  f16  2MB
constexpr size_t OFF_WV    = 2 * MB;
constexpr size_t OFF_WR    = 4 * MB;
constexpr size_t OFF_WO    = 6 * MB;
constexpr size_t OFF_WCK   = 8 * MB;     // [3C,C] f16  6MB
constexpr size_t OFF_WCV   = 14 * MB;
constexpr size_t OFF_WCW   = 20 * MB;    // [C,3C] f16  6MB
constexpr size_t OFF_WCR   = 26 * MB;    // 2MB -> 28MB
constexpr size_t OFF_WFULL = 28 * MB;    // [H,T,T] f16 32MB -> 60MB
constexpr size_t OFF_ACT   = 60 * MB;    // [BT,C] f16 16MB (xs1, then xs2)
constexpr size_t OFF_K     = 76 * MB;    // [BT,C] f32 32MB (exp(k))
constexpr size_t OFF_KV    = 108 * MB;   // [BT,C] f16 16MB
constexpr size_t OFF_SIGR  = 124 * MB;   // [BT,C] f32 32MB (sig(r), later sig(r2))
constexpr size_t OFF_SUMK  = 156 * MB;   // [BT,C] f32 32MB
constexpr size_t OFF_Y     = 188 * MB;   // [BT,C] f16 16MB
constexpr size_t OFF_X1    = 204 * MB;   // [BT,C] f32 32MB
constexpr size_t OFF_G     = 76 * MB;    // [BT,3C] f16 48MB (reuses K+KV after phase 6)
constexpr size_t OFF_NORM  = 236 * MB;   // [BT] f32

// ---------------------------------------------------------------- WMMA helpers
__device__ __forceinline__ v8f zero8() {
  v8f r;
#pragma unroll
  for (int i = 0; i < 8; ++i) r[i] = 0.0f;
  return r;
}

__device__ __forceinline__ v8f wmma_f16(v16h a, v16h b, v8f c) {
  return __builtin_amdgcn_wmma_f32_16x16x32_f16(false, a, false, b, (short)0, c, false, false);
}

// A fragment 16x32 f16: lane group g holds K = g*16 .. +15 (contiguous run)
__device__ __forceinline__ v16h load_fragA(const half_t* A, int lda, int row0, int k0, int lane) {
  int r = lane & 15, g = (lane >> 4) & 1;
  const half_t* p = A + (size_t)(row0 + r) * lda + (k0 + g * 16);
  union { v16h v; v8h h[2]; } u;
  u.h[0] = *(const v8h*)p;
  u.h[1] = *(const v8h*)(p + 8);
  return u.v;
}

// B fragment 32x16 f16, B[k][n] = Bw[n][k] (weights stored [N,K] row-major):
// lane group g holds K = k0+g*16 .. +15 for column (lane&15) -> contiguous run.
__device__ __forceinline__ v16h load_fragB(const half_t* Bw, int ldb, int col0, int k0, int lane) {
  const half_t* p = Bw + (size_t)(col0 + (lane & 15)) * ldb + (k0 + ((lane >> 4) & 1) * 16);
  union { v16h v; v8h h[2]; } u;
  u.h[0] = *(const v8h*)p;
  u.h[1] = *(const v8h*)(p + 8);
  return u.v;
}

// Strided B fragment: B[k][n] = src[k*strideK + n]  (kv tensor, K-major stride C)
__device__ __forceinline__ v16h load_fragB_strided(const half_t* src, int strideK, int col0, int k0, int lane) {
  int col = col0 + (lane & 15);
  int kk = k0 + ((lane >> 4) & 1) * 16;
  v16h f;
#pragma unroll
  for (int i = 0; i < 16; ++i) f[i] = src[(size_t)(kk + i) * strideK + col];
  return f;
}

// Core: wave computes a 32 x (16*NF) tile.  A:[M,K] f16 row-major, Bw:[N,K]
// f16 row-major.  Unroll-2 lets the scheduler overlap the next k-step's loads
// with this step's WMMA chain using fresh registers (no manual double buffer).
template <int NF>
__device__ __forceinline__ void gemm_acc(const half_t* A, const half_t* Bw, int K, int lda, int ldb,
                                         int row0, int col0, v8f acc[2][NF]) {
  int lane = threadIdx.x & 31;
#pragma unroll 2
  for (int k0 = 0; k0 < K; k0 += 32) {
    __builtin_prefetch(A + (size_t)(row0 + (lane & 15)) * lda + k0 + 64, 0, 3);
    v16h a0 = load_fragA(A, lda, row0, k0, lane);
    v16h a1 = load_fragA(A, lda, row0 + 16, k0, lane);
    v16h b[NF];
#pragma unroll
    for (int j = 0; j < NF; ++j) b[j] = load_fragB(Bw, ldb, col0 + 16 * j, k0, lane);
#pragma unroll
    for (int j = 0; j < NF; ++j) {
      acc[0][j] = wmma_f16(a0, b[j], acc[0][j]);
      acc[1][j] = wmma_f16(a1, b[j], acc[1][j]);
    }
  }
}

// ---------------------------------------------------------------- elementwise kernels
__global__ void k_f2h(const float* __restrict__ src, half_t* __restrict__ dst, int n) {
  int i = blockIdx.x * 256 + threadIdx.x;
  if (i < n) dst[i] = (half_t)src[i];
}

// w[h,t,u] = (u<=t) ? time_w[h, TT-1+u-t]*alpha[h,u]*beta[h,t] : 0
__global__ void k_wfull(const float* __restrict__ tw, const float* __restrict__ ta,
                        const float* __restrict__ tb, half_t* __restrict__ wf) {
  size_t e = (size_t)blockIdx.x * 256 + threadIdx.x;
  int h = (int)(e >> 20);
  int rest = (int)(e & ((1u << 20) - 1));
  int t = rest >> 10, u = rest & 1023;
  float val = 0.0f;
  if (u <= t) val = tw[h * TT_ + (TT_ - 1) + u - t] * ta[h * TT_ + u] * tb[h * TT_ + t];
  wf[e] = (half_t)val;
}

// inv_norm[row] = 1/(||x_row|| * C^-0.5 + 1e-12)
__global__ void k_rownorm(const float* __restrict__ src, float* __restrict__ inv) {
  __shared__ float red[256];
  int row = blockIdx.x, tid = threadIdx.x;
  float s = 0.0f;
  for (int c = tid; c < C_; c += 256) {
    float v = src[(size_t)row * C_ + c];
    s += v * v;
  }
  red[tid] = s;
  __syncthreads();
  for (int off = 128; off > 0; off >>= 1) {
    if (tid < off) red[tid] += red[tid + off];
    __syncthreads();
  }
  if (tid == 0) inv[row] = 1.0f / (sqrtf(red[0]) * 0.03125f + 1e-12f);
}

// xs = time_shift(fixed_norm(src)) -> f16
__global__ void k_shift(const float* __restrict__ src, const float* __restrict__ inv,
                        half_t* __restrict__ dst) {
  size_t e = (size_t)blockIdx.x * 256 + threadIdx.x;
  size_t row = e >> 10;
  int c = (int)(e & 1023);
  int t = (int)(row & 1023);
  float val = 0.0f;
  if (c < C_ / 2) {
    if (t > 0) val = src[(row - 1) * C_ + c] * inv[row - 1];
  } else {
    val = src[row * C_ + c] * inv[row];
  }
  dst[e] = (half_t)val;
}

// sum_k[b,t,c] = cumsum over t of exp(k)
__global__ void k_cumsum(const float* __restrict__ kin, float* __restrict__ sout) {
  int b = blockIdx.y;
  int c = blockIdx.x * 256 + threadIdx.x;
  float run = 0.0f;
  for (int t = 0; t < T_; ++t) {
    size_t idx = ((size_t)b * T_ + t) * C_ + c;
    run += kin[idx];
    sout[idx] = run;
  }
}

// ---------------------------------------------------------------- GEMM kernels (128x128 block, 8 waves)
#define GEMM_PROLOG(BN_WAVE)                                          \
  int wave = threadIdx.x >> 5;                                        \
  int wy = wave >> 1, wx = wave & 1;                                  \
  int row0 = blockIdx.y * 128 + wy * 32;                              \
  int col0 = blockIdx.x * (2 * (BN_WAVE)) + wx * (BN_WAVE);           \
  int lane = threadIdx.x & 31;                                        \
  int nl = lane & 15, mo = ((lane >> 4) & 1) * 8;

__global__ void k_gemm_exp(const half_t* __restrict__ A, const half_t* __restrict__ Bw,
                           const float* __restrict__ bias, float* __restrict__ out,
                           int N, int K) {
  GEMM_PROLOG(64)
  v8f acc[2][4];
#pragma unroll
  for (int i = 0; i < 2; ++i)
#pragma unroll
    for (int j = 0; j < 4; ++j) acc[i][j] = zero8();
  gemm_acc<4>(A, Bw, K, K, K, row0, col0, acc);
#pragma unroll
  for (int mi = 0; mi < 2; ++mi)
#pragma unroll
    for (int j = 0; j < 4; ++j) {
      int n = col0 + 16 * j + nl;
      float bia = bias[n];
#pragma unroll
      for (int v = 0; v < 8; ++v) {
        int m = row0 + mi * 16 + mo + v;
        out[(size_t)m * N + n] = expf(acc[mi][j][v] + bia);
      }
    }
}

__global__ void k_gemm_kv(const half_t* __restrict__ A, const half_t* __restrict__ Bw,
                          const float* __restrict__ bias, const float* __restrict__ kbuf,
                          half_t* __restrict__ kvout, int N, int K) {
  GEMM_PROLOG(64)
  v8f acc[2][4];
#pragma unroll
  for (int i = 0; i < 2; ++i)
#pragma unroll
    for (int j = 0; j < 4; ++j) acc[i][j] = zero8();
  gemm_acc<4>(A, Bw, K, K, K, row0, col0, acc);
#pragma unroll
  for (int mi = 0; mi < 2; ++mi)
#pragma unroll
    for (int j = 0; j < 4; ++j) {
      int n = col0 + 16 * j + nl;
      float bia = bias[n];
#pragma unroll
      for (int v = 0; v < 8; ++v) {
        int m = row0 + mi * 16 + mo + v;
        size_t idx = (size_t)m * N + n;
        kvout[idx] = (half_t)(kbuf[idx] * (acc[mi][j][v] + bia));
      }
    }
}

__global__ void k_gemm_sig(const half_t* __restrict__ A, const half_t* __restrict__ Bw,
                           const float* __restrict__ bias, float* __restrict__ out,
                           int N, int K) {
  GEMM_PROLOG(64)
  v8f acc[2][4];
#pragma unroll
  for (int i = 0; i < 2; ++i)
#pragma unroll
    for (int j = 0; j < 4; ++j) acc[i][j] = zero8();
  gemm_acc<4>(A, Bw, K, K, K, row0, col0, acc);
#pragma unroll
  for (int mi = 0; mi < 2; ++mi)
#pragma unroll
    for (int j = 0; j < 4; ++j) {
      int n = col0 + 16 * j + nl;
      float bia = bias[n];
#pragma unroll
      for (int v = 0; v < 8; ++v) {
        int m = row0 + mi * 16 + mo + v;
        out[(size_t)m * N + n] = 1.0f / (1.0f + expf(-(acc[mi][j][v] + bia)));
      }
    }
}

// x1 = x + (y @ Wo^T + bo) * gamma[t]
__global__ void k_gemm_out(const half_t* __restrict__ A, const half_t* __restrict__ Bw,
                           const float* __restrict__ bias, const float* __restrict__ xin,
                           const float* __restrict__ gamma, float* __restrict__ x1,
                           int N, int K) {
  GEMM_PROLOG(64)
  v8f acc[2][4];
#pragma unroll
  for (int i = 0; i < 2; ++i)
#pragma unroll
    for (int j = 0; j < 4; ++j) acc[i][j] = zero8();
  gemm_acc<4>(A, Bw, K, K, K, row0, col0, acc);
#pragma unroll
  for (int mi = 0; mi < 2; ++mi)
#pragma unroll
    for (int j = 0; j < 4; ++j) {
      int n = col0 + 16 * j + nl;
      float bia = bias[n];
#pragma unroll
      for (int v = 0; v < 8; ++v) {
        int m = row0 + mi * 16 + mo + v;
        int t = m & (T_ - 1);
        size_t idx = (size_t)m * N + n;
        x1[idx] = xin[idx] + (acc[mi][j][v] + bia) * gamma[t];
      }
    }
}

// out = x1 + sig_r2 * (g @ Wcw^T + bcw)
__global__ void k_gemm_final(const half_t* __restrict__ A, const half_t* __restrict__ Bw,
                             const float* __restrict__ bias, const float* __restrict__ x1,
                             const float* __restrict__ sigr, float* __restrict__ out,
                             int N, int K) {
  GEMM_PROLOG(64)
  v8f acc[2][4];
#pragma unroll
  for (int i = 0; i < 2; ++i)
#pragma unroll
    for (int j = 0; j < 4; ++j) acc[i][j] = zero8();
  gemm_acc<4>(A, Bw, K, K, K, row0, col0, acc);
#pragma unroll
  for (int mi = 0; mi < 2; ++mi)
#pragma unroll
    for (int j = 0; j < 4; ++j) {
      int n = col0 + 16 * j + nl;
      float bia = bias[n];
#pragma unroll
      for (int v = 0; v < 8; ++v) {
        int m = row0 + mi * 16 + mo + v;
        size_t idx = (size_t)m * N + n;
        out[idx] = x1[idx] + sigr[idx] * (acc[mi][j][v] + bia);
      }
    }
}

// Fused dual GEMM: k2 = xs@Wck^T+bck, v2 = xs@Wcv^T+bcv, g = mish(k2)*v2 (f16).
// A fragments shared; B panels processed sequentially to bound register pressure.
__global__ void k_gemm_dual(const half_t* __restrict__ A, const half_t* __restrict__ B1,
                            const half_t* __restrict__ B2, const float* __restrict__ bias1,
                            const float* __restrict__ bias2, half_t* __restrict__ gout,
                            int N, int K) {
  GEMM_PROLOG(64)
  v8f ak[2][4], av[2][4];
#pragma unroll
  for (int i = 0; i < 2; ++i)
#pragma unroll
    for (int j = 0; j < 4; ++j) { ak[i][j] = zero8(); av[i][j] = zero8(); }
#pragma unroll 2
  for (int k0 = 0; k0 < K; k0 += 32) {
    v16h a0 = load_fragA(A, K, row0, k0, lane);
    v16h a1 = load_fragA(A, K, row0 + 16, k0, lane);
    v16h b[4];
#pragma unroll
    for (int j = 0; j < 4; ++j) b[j] = load_fragB(B1, K, col0 + 16 * j, k0, lane);
#pragma unroll
    for (int j = 0; j < 4; ++j) {
      ak[0][j] = wmma_f16(a0, b[j], ak[0][j]);
      ak[1][j] = wmma_f16(a1, b[j], ak[1][j]);
    }
#pragma unroll
    for (int j = 0; j < 4; ++j) b[j] = load_fragB(B2, K, col0 + 16 * j, k0, lane);
#pragma unroll
    for (int j = 0; j < 4; ++j) {
      av[0][j] = wmma_f16(a0, b[j], av[0][j]);
      av[1][j] = wmma_f16(a1, b[j], av[1][j]);
    }
  }
#pragma unroll
  for (int mi = 0; mi < 2; ++mi)
#pragma unroll
    for (int j = 0; j < 4; ++j) {
      int n = col0 + 16 * j + nl;
      float bb1 = bias1[n], bb2 = bias2[n];
#pragma unroll
      for (int v = 0; v < 8; ++v) {
        int m = row0 + mi * 16 + mo + v;
        float kk = ak[mi][j][v] + bb1;
        float vv = av[mi][j][v] + bb2;
        float mish = kk * tanhf(log1pf(expf(kk)));
        gout[(size_t)m * N + n] = (half_t)(mish * vv);
      }
    }
}

// Batched triangular einsum: per (b,h): wkv[t,c] = sum_u w[h,t,u]*kv[b,u,h,c];
// epilogue y = sig(r)*wkv/sum_k -> f16.  Block 128x64, wave tile 32x32 (NF=2).
// K loop trimmed to the wave's own row block (w is lower triangular).
__global__ void k_wkv(const half_t* __restrict__ wfull, const half_t* __restrict__ kvh,
                      const float* __restrict__ sigr, const float* __restrict__ sumk,
                      half_t* __restrict__ yh) {
  int bh = blockIdx.z;
  int h = bh & (H_ - 1);
  int b = bh >> 4;
  const half_t* A = wfull + (size_t)h * T_ * T_;
  const half_t* Bs = kvh + (size_t)b * T_ * C_ + h * HS_;

  int wave = threadIdx.x >> 5;
  int wy = wave >> 1, wx = wave & 1;
  int row0 = blockIdx.y * 128 + wy * 32;
  int col0 = wx * 32;
  int lane = threadIdx.x & 31;
  int nl = lane & 15, mo = ((lane >> 4) & 1) * 8;

  v8f acc[2][2];
#pragma unroll
  for (int i = 0; i < 2; ++i)
#pragma unroll
    for (int j = 0; j < 2; ++j) acc[i][j] = zero8();

  int kmax = row0 + 32;  // u > t contributes 0
  for (int k0 = 0; k0 < kmax; k0 += 32) {
    v16h a0 = load_fragA(A, T_, row0, k0, lane);
    v16h a1 = load_fragA(A, T_, row0 + 16, k0, lane);
    v16h bfr[2];
#pragma unroll
    for (int j = 0; j < 2; ++j) bfr[j] = load_fragB_strided(Bs, C_, col0 + 16 * j, k0, lane);
#pragma unroll
    for (int j = 0; j < 2; ++j) {
      acc[0][j] = wmma_f16(a0, bfr[j], acc[0][j]);
      acc[1][j] = wmma_f16(a1, bfr[j], acc[1][j]);
    }
  }

#pragma unroll
  for (int mi = 0; mi < 2; ++mi)
#pragma unroll
    for (int j = 0; j < 2; ++j) {
      int cc = col0 + 16 * j + nl;
#pragma unroll
      for (int v = 0; v < 8; ++v) {
        int t = row0 + mi * 16 + mo + v;
        size_t idx = ((size_t)b * T_ + t) * C_ + h * HS_ + cc;
        yh[idx] = (half_t)(sigr[idx] * acc[mi][j][v] / sumk[idx]);
      }
    }
}

// ---------------------------------------------------------------- launcher
extern "C" void kernel_launch(void* const* d_in, const int* in_sizes, int n_in,
                              void* d_out, int out_size, void* d_ws, size_t ws_size,
                              hipStream_t stream) {
  (void)in_sizes; (void)n_in; (void)out_size; (void)ws_size;

  const float* x     = (const float*)d_in[0];
  const float* tw    = (const float*)d_in[1];
  const float* ta    = (const float*)d_in[2];
  const float* tb    = (const float*)d_in[3];
  const float* gamma = (const float*)d_in[4];
  const float* Wk  = (const float*)d_in[5];   const float* bk  = (const float*)d_in[6];
  const float* Wv  = (const float*)d_in[7];   const float* bv  = (const float*)d_in[8];
  const float* Wr  = (const float*)d_in[9];   const float* br  = (const float*)d_in[10];
  const float* Wo  = (const float*)d_in[11];  const float* bo  = (const float*)d_in[12];
  const float* Wck = (const float*)d_in[13];  const float* bck = (const float*)d_in[14];
  const float* Wcv = (const float*)d_in[15];  const float* bcv = (const float*)d_in[16];
  const float* Wcw = (const float*)d_in[17];  const float* bcw = (const float*)d_in[18];
  const float* Wcr = (const float*)d_in[19];  const float* bcr = (const float*)d_in[20];

  char* ws = (char*)d_ws;
  half_t* WkH   = (half_t*)(ws + OFF_WK);
  half_t* WvH   = (half_t*)(ws + OFF_WV);
  half_t* WrH   = (half_t*)(ws + OFF_WR);
  half_t* WoH   = (half_t*)(ws + OFF_WO);
  half_t* WckH  = (half_t*)(ws + OFF_WCK);
  half_t* WcvH  = (half_t*)(ws + OFF_WCV);
  half_t* WcwH  = (half_t*)(ws + OFF_WCW);
  half_t* WcrH  = (half_t*)(ws + OFF_WCR);
  half_t* Wfull = (half_t*)(ws + OFF_WFULL);
  half_t* act   = (half_t*)(ws + OFF_ACT);
  float*  Kbuf  = (float*)(ws + OFF_K);
  half_t* KVh   = (half_t*)(ws + OFF_KV);
  float*  SigR  = (float*)(ws + OFF_SIGR);
  float*  SumK  = (float*)(ws + OFF_SUMK);
  half_t* Yh    = (half_t*)(ws + OFF_Y);
  float*  X1    = (float*)(ws + OFF_X1);
  half_t* Gh    = (half_t*)(ws + OFF_G);
  float*  Norms = (float*)(ws + OFF_NORM);
  float*  Out   = (float*)d_out;

  dim3 blk(256);

  // Phase 0: weights -> f16, build Toeplitz w
  k_f2h<<<(C_ * C_) / 256, blk, 0, stream>>>(Wk, WkH, C_ * C_);
  k_f2h<<<(C_ * C_) / 256, blk, 0, stream>>>(Wv, WvH, C_ * C_);
  k_f2h<<<(C_ * C_) / 256, blk, 0, stream>>>(Wr, WrH, C_ * C_);
  k_f2h<<<(C_ * C_) / 256, blk, 0, stream>>>(Wo, WoH, C_ * C_);
  k_f2h<<<(C3_ * C_) / 256, blk, 0, stream>>>(Wck, WckH, C3_ * C_);
  k_f2h<<<(C3_ * C_) / 256, blk, 0, stream>>>(Wcv, WcvH, C3_ * C_);
  k_f2h<<<(C_ * C3_) / 256, blk, 0, stream>>>(Wcw, WcwH, C_ * C3_);
  k_f2h<<<(C_ * C_) / 256, blk, 0, stream>>>(Wcr, WcrH, C_ * C_);
  k_wfull<<<(H_ * T_ * T_) / 256, blk, 0, stream>>>(tw, ta, tb, Wfull);

  // Phase 1: xs1 = time_shift(fixed_norm(x)) -> f16
  k_rownorm<<<BT_, blk, 0, stream>>>(x, Norms);
  k_shift<<<(BT_ * C_) / 256, blk, 0, stream>>>(x, Norms, act);

  dim3 g8x64(C_ / 128, BT_ / 128);   // N=1024 tiles x M=8192 tiles

  // Phases 2-4: k (exp), kv (k*v f16), sigmoid(r)
  k_gemm_exp<<<g8x64, blk, 0, stream>>>(act, WkH, bk, Kbuf, C_, C_);
  k_gemm_kv <<<g8x64, blk, 0, stream>>>(act, WvH, bv, Kbuf, KVh, C_, C_);
  k_gemm_sig<<<g8x64, blk, 0, stream>>>(act, WrH, br, SigR, C_, C_);

  // Phase 5: cumsum of exp(k) over time
  k_cumsum<<<dim3(C_ / 256, Bb_), blk, 0, stream>>>(Kbuf, SumK);

  // Phase 6: triangular batched einsum + epilogue -> y f16
  k_wkv<<<dim3(1, T_ / 128, Bb_ * H_), blk, 0, stream>>>(Wfull, KVh, SigR, SumK, Yh);

  // Phase 7: x1 = x + (y@Wo^T + bo)*gamma
  k_gemm_out<<<g8x64, blk, 0, stream>>>(Yh, WoH, bo, x, gamma, X1, C_, C_);

  // Phase 8: xs2 = time_shift(fixed_norm(x1)) -> f16 (reuses act)
  k_rownorm<<<BT_, blk, 0, stream>>>(X1, Norms);
  k_shift<<<(BT_ * C_) / 256, blk, 0, stream>>>(X1, Norms, act);

  // Phase 9: fused dual GEMM -> g = mish(k2)*v2 (f16, [BT,3C])
  k_gemm_dual<<<dim3(C3_ / 128, BT_ / 128), blk, 0, stream>>>(act, WckH, WcvH, bck, bcv, Gh, C3_, C_);

  // Phase 10: sigmoid(r2)
  k_gemm_sig<<<g8x64, blk, 0, stream>>>(act, WcrH, bcr, SigR, C_, C_);

  // Phase 11: out = x1 + sig(r2)*(g@Wcw^T + bcw)
  k_gemm_final<<<g8x64, blk, 0, stream>>>(Gh, WcwH, bcw, X1, SigR, Out, C_, C3_);
}
